// MergingBlock_39960375722526
// MI455X (gfx1250) — matrix operand order repaired and checked by reading, not compiled, not run on hardware
//
#include <hip/hip_runtime.h>

// ---------------------------------------------------------------------------
// MergingBlock on gfx1250: fused implicit-GEMM conv3x3 (520->256, f16 WMMA)
// + PReLU + 1x1 conv (256->36, f16 WMMA) + dual softmax(18) + 3x3 unfold merge.
// Round 3: fragment loads as ds_load_b128 pairs (table-order k lane layout);
// double-buffered LDS pipeline; weight tiles staged via tensor_load_to_lds.
// ---------------------------------------------------------------------------

#define USE_TDM 1

typedef __attribute__((ext_vector_type(16))) _Float16     v16h;
typedef __attribute__((ext_vector_type(8)))  float        v8f;
typedef __attribute__((ext_vector_type(4)))  unsigned int u32x4;
typedef __attribute__((ext_vector_type(8)))  unsigned int u32x8;

#define B_    2
#define H_    256
#define W_    448
#define HW_   (H_ * W_)
#define MTOT  (B_ * HW_)          // 229376
#define CIN   520
#define HID   256
#define KREAL (CIN * 9)           // 4680
#define KPAD  4704                // 147 * 32
#define NSTEP (KPAD / 32)         // 147
#define TM    128                 // pixels per workgroup
#define N2    48                  // padded conv2 output channels
#define COUT  36

union Frag { v16h h; u32x4 q[2]; };

// A-matrix 16x32 f16 fragment (ISA 7.12.2): lanes 0-15 M=lane; VGPR v holds
// K pair at K = 2v + (v>=4 ? 8 : 0) + 8*(lane>=16).
// Within a lane that is two contiguous 16B runs: K+0..7 and K+16..23.
__device__ __forceinline__ v16h load_a_frag(const _Float16* p, int row0, int ld, int k0) {
    int lane = threadIdx.x & 31;
    int lr = lane & 15, hi = lane >> 4;
    const _Float16* base = p + (size_t)(row0 + lr) * ld + k0 + 8 * hi;
    Frag f;
    f.q[0] = *(const u32x4*)(base);        // K +0 .. +7
    f.q[1] = *(const u32x4*)(base + 16);   // K +16 .. +23
    return f.h;
}

// B-matrix 32x16 f16 fragment; source stored N-major (row n = K run).
// lane%16 = N column; same K order as A: K+0..7 and K+16..23, offset 8*(lane>=16).
__device__ __forceinline__ v16h load_b_frag(const _Float16* p, int n0, int ld, int k0) {
    int lane = threadIdx.x & 31;
    int lr = lane & 15, hi = lane >> 4;
    const _Float16* base = p + (size_t)(n0 + lr) * ld + k0 + 8 * hi;
    Frag f;
    f.q[0] = *(const u32x4*)(base);        // K +0 .. +7
    f.q[1] = *(const u32x4*)(base + 16);   // K +16 .. +23
    return f.h;
}

#if USE_TDM
// Issue a TDM load of a 32(k) x 256(n) f16 tile of the [256][KPAD] weight
// tensor into LDS at byte offset lds_off (tile rows land contiguously: [n][32]).
// D# per ISA 8.3/8.4: group0 = {count, lds_addr, global_addr, type=2},
// group1 = {mask=0, data_size=2B, dims/strides, tile dims}.
__device__ __forceinline__ void tdm_issue_b(const _Float16* gtile, unsigned lds_off) {
    unsigned long long a = (unsigned long long)(uintptr_t)gtile;
    u32x4 g0;
    g0.x = 1u;                                   // count=1, user desc
    g0.y = lds_off;                              // lds_addr
    g0.z = (unsigned)a;                          // global_addr[31:0]
    g0.w = (unsigned)(a >> 32) | 0x80000000u;    // global_addr[56:32] | type=2
    u32x8 g1;
    g1.s0 = 0x00010000u;     // workgroup_mask=0, data_size=1 (2 bytes)
    g1.s1 = 0x12600000u;     // tensor_dim0[15:0]=4704 @ bits 63:48
    g1.s2 = 0x01000000u;     // tensor_dim0[31:16]=0, tensor_dim1[15:0]=256
    g1.s3 = 0x00200000u;     // tensor_dim1[31:16]=0, tile_dim0=32
    g1.s4 = 0x00000100u;     // tile_dim1=256, tile_dim2=0
    g1.s5 = 0x00001260u;     // tensor_dim0_stride[31:0]=4704
    g1.s6 = 0u;              // stride hi, tensor_dim1_stride lo
    g1.s7 = 0u;
    asm volatile("tensor_load_to_lds %0, %1" :: "s"(g0), "s"(g1) : "memory");
}
#endif

// --- weight packing -------------------------------------------------------

// w1: [HID][CIN][3][3] f32 (OIHW) -> Wt1 f16 [HID][KPAD], k = kpos*CIN + c
__global__ void pack_w1_kernel(const float* __restrict__ w1, _Float16* __restrict__ Wt1) {
    int idx = blockIdx.x * 256 + threadIdx.x;
    if (idx >= HID * KPAD) return;
    int n = idx / KPAD;
    int k = idx - n * KPAD;
    float v = 0.f;
    if (k < KREAL) {
        int kpos = k / CIN;
        int c    = k - kpos * CIN;
        int ky = kpos / 3, kx = kpos % 3;
        v = w1[((n * CIN + c) * 3 + ky) * 3 + kx];
    }
    Wt1[idx] = (_Float16)v;
}

// w2: [COUT][HID][1][1] f32 -> Wt2 f16 [N2][HID] (rows 36..47 zero)
__global__ void pack_w2_kernel(const float* __restrict__ w2, _Float16* __restrict__ Wt2) {
    int idx = blockIdx.x * 256 + threadIdx.x;
    if (idx >= N2 * HID) return;
    int n = idx / HID;
    int k = idx - n * HID;
    Wt2[idx] = (_Float16)((n < COUT) ? w2[n * HID + k] : 0.f);
}

// --- fused main kernel ----------------------------------------------------
// LDS layout (dynamic, 136 KB), double-buffered A/B tiles:
//   [0      , 24576) : W2  f16 [48][256]
//   [24576  , 90112) : X   f16 [128][256]
//   [90112  ,106496) : A0/A1 f16 [128][32] x2        } reused in phase-2 as
//   [106496 ,139264) : B0/B1 f16 [256][32] x2        } mask f32 [128][36]
#define LDS_W2    0
#define LDS_X     24576
#define LDS_AB    (24576 + 65536)
#define LDS_BOFF  (LDS_AB + 2 * 8192)
#define LDS_TOTAL (24576 + 65536 + 2 * 8192 + 2 * 16384)

__global__ __launch_bounds__(256)
void merging_main_kernel(const float* __restrict__ feature,
                         const float* __restrict__ init_flow,
                         const float* __restrict__ flow_fix,
                         const float* __restrict__ b1,
                         const float* __restrict__ prelu_w,
                         const float* __restrict__ b2,
                         const _Float16* __restrict__ Wt1,
                         const _Float16* __restrict__ Wt2,
                         float* __restrict__ out) {
    extern __shared__ char smem[];
    _Float16* sW2 = (_Float16*)(smem + LDS_W2);
    _Float16* sX  = (_Float16*)(smem + LDS_X);
    float*    sM  = (float*)(smem + LDS_AB);   // reused after GEMM1

    const int tid  = threadIdx.x;
    const int m0   = blockIdx.x * TM;
    const int lane = tid & 31;
    const int lr   = lane & 15;
    const int hi   = lane >> 4;
    const int wave = tid >> 5;
    const int wm   = wave & 3;   // M quadrant (32 rows)
    const int wn   = wave >> 2;  // N half (128 cols)

    // Stage W2 (48x256 f16 = 24 KB) into LDS with 128b copies.
    {
        const uint4* src = (const uint4*)Wt2;
        uint4*       dst = (uint4*)sW2;
        for (int i = tid; i < (N2 * HID * 2) / 16; i += 256) dst[i] = src[i];
    }

    // Per-thread im2col coordinates: thread handles half a row of the A tile.
    const int mr     = tid >> 1;
    const int kkbase = (tid & 1) * 16;
    const int mg     = m0 + mr;
    const int bb     = mg / HW_;
    const int remp   = mg - bb * HW_;
    const int py     = remp / W_;
    const int px     = remp - py * W_;

    auto gather_a = [&](int k0, _Float16* dst) {
#pragma unroll 4
        for (int e = 0; e < 16; ++e) {
            int k = k0 + kkbase + e;
            float val = 0.f;
            if (k < KREAL) {
                int kpos = k / CIN;
                int c    = k - kpos * CIN;
                int yy = py + kpos / 3 - 1;
                int xx = px + kpos % 3 - 1;
                if (yy >= 0 && yy < H_ && xx >= 0 && xx < W_) {
                    if (c < 4)
                        val = init_flow[((bb * 4 + c) * H_ + yy) * W_ + xx];
                    else if (c < 8)
                        val = flow_fix[((bb * 4 + (c - 4)) * H_ + yy) * W_ + xx];
                    else
                        val = feature[((bb * 512 + (c - 8)) * H_ + yy) * W_ + xx];
                }
            }
            dst[mr * 32 + kkbase + e] = (_Float16)val;
        }
    };

#if !USE_TDM
    auto copy_b = [&](int k0, _Float16* dst) {
        const uint4* src = (const uint4*)(Wt1 + (size_t)tid * KPAD + k0);
        uint4*       d   = (uint4*)(dst + tid * 32);
#pragma unroll
        for (int i = 0; i < 4; ++i) d[i] = src[i];
    };
#endif

    v8f acc[2][8];
    {
        v8f z = {};
#pragma unroll
        for (int i = 0; i < 2; ++i)
#pragma unroll
            for (int j = 0; j < 8; ++j) acc[i][j] = z;
    }

    // ---------------- GEMM1: conv3x3 implicit GEMM (double-buffered) ------
    // Prologue: fill stage 0.
    gather_a(0, (_Float16*)(smem + LDS_AB));
#if USE_TDM
    if (wave == 0) {
        tdm_issue_b(Wt1, LDS_BOFF);
        __builtin_amdgcn_s_wait_tensorcnt(0);
    }
#else
    copy_b(0, (_Float16*)(smem + LDS_BOFF));
#endif
    __syncthreads();

    for (int step = 0; step < NSTEP; ++step) {
        const int cur = step & 1;
        const int nxt = cur ^ 1;
        _Float16* sAc = (_Float16*)(smem + LDS_AB   + cur * 8192);
        _Float16* sBc = (_Float16*)(smem + LDS_BOFF + cur * 16384);

        // Kick off next stage first so it overlaps the WMMA burst below.
        if (step + 1 < NSTEP) {
            int k0n = (step + 1) * 32;
#if USE_TDM
            if (wave == 0)
                tdm_issue_b(Wt1 + k0n, LDS_BOFF + nxt * 16384);
#else
            copy_b(k0n, (_Float16*)(smem + LDS_BOFF + nxt * 16384));
#endif
            gather_a(k0n, (_Float16*)(smem + LDS_AB + nxt * 8192));
        }

        // Compute on the current stage.
        v16h a0 = load_a_frag(sAc, wm * 32,      32, 0);
        v16h a1 = load_a_frag(sAc, wm * 32 + 16, 32, 0);
#pragma unroll
        for (int ni = 0; ni < 8; ++ni) {
            v16h bf = load_b_frag(sBc, wn * 128 + ni * 16, 32, 0);
            acc[0][ni] = __builtin_amdgcn_wmma_f32_16x16x32_f16(
                false, a0, false, bf, (short)0, acc[0][ni], false, false);
            acc[1][ni] = __builtin_amdgcn_wmma_f32_16x16x32_f16(
                false, a1, false, bf, (short)0, acc[1][ni], false, false);
        }

#if USE_TDM
        if (wave == 0 && step + 1 < NSTEP)
            __builtin_amdgcn_s_wait_tensorcnt(0);   // next B tile landed
#endif
        __syncthreads();
    }

    // Epilogue: bias + PReLU, spill x tile to LDS as f16.
#pragma unroll
    for (int ni = 0; ni < 8; ++ni) {
        int n = wn * 128 + ni * 16 + lr;
        float bias = b1[n];
        float pw   = prelu_w[n];
#pragma unroll
        for (int mi = 0; mi < 2; ++mi) {
#pragma unroll
            for (int r = 0; r < 8; ++r) {
                int mm = wm * 32 + mi * 16 + hi * 8 + r;
                float v = acc[mi][ni][r] + bias;
                v = (v >= 0.f) ? v : pw * v;
                sX[mm * 256 + n] = (_Float16)v;
            }
        }
    }
    __syncthreads();

    // ---------------- GEMM2: 1x1 conv 256 -> 36 (padded 48) --------------
#pragma unroll
    for (int t3 = 0; t3 < 3; ++t3) {
        v8f acc2 = {};
#pragma unroll
        for (int k0 = 0; k0 < HID; k0 += 32) {
            v16h a = load_a_frag(sX,  wave * 16, 256, k0);
            v16h b = load_b_frag(sW2, t3 * 16,   256, k0);
            acc2 = __builtin_amdgcn_wmma_f32_16x16x32_f16(
                false, a, false, b, (short)0, acc2, false, false);
        }
        int n = t3 * 16 + lr;
        if (n < COUT) {
            float bias = b2[n];
#pragma unroll
            for (int r = 0; r < 8; ++r) {
                int mm = wave * 16 + hi * 8 + r;
                sM[mm * COUT + n] = acc2[r] + bias;
            }
        }
    }
    __syncthreads();

    // ---------------- softmax(18) x2 + 3x3 unfold merge ------------------
    if (tid < TM) {
        int p  = tid;
        int gm = m0 + p;
        int b  = gm / HW_;
        int rr = gm - b * HW_;
        int yy = rr / W_;
        int xx = rr - yy * W_;

        float mv[36];
#pragma unroll
        for (int i = 0; i < 36; ++i) mv[i] = sM[p * COUT + i];

#pragma unroll
        for (int h = 0; h < 2; ++h) {
            float* m = mv + h * 18;
            float mx = m[0];
#pragma unroll
            for (int i = 1; i < 18; ++i) mx = fmaxf(mx, m[i]);
            float s = 0.f;
#pragma unroll
            for (int i = 0; i < 18; ++i) { m[i] = __expf(m[i] - mx); s += m[i]; }
            float inv = 1.f / s;

            float o0 = 0.f, o1 = 0.f;
            int c0 = h * 2;
#pragma unroll
            for (int j = 0; j < 9; ++j) {
                int yv = yy + j / 3 - 1;
                int xv = xx + j % 3 - 1;
                if (yv >= 0 && yv < H_ && xv >= 0 && xv < W_) {
                    float w0 = m[j] * inv, w1 = m[9 + j] * inv;
                    int base0 = ((b * 4 + c0) * H_ + yv) * W_ + xv;
                    int base1 = ((b * 4 + c0 + 1) * H_ + yv) * W_ + xv;
                    o0 += w0 * init_flow[base0] + w1 * flow_fix[base0];
                    o1 += w0 * init_flow[base1] + w1 * flow_fix[base1];
                }
            }
            int ob = h * B_ + b;   // batch slot in [2B,2,H,W]
            out[((ob * 2 + 0) * H_ + yy) * W_ + xx] = o0;
            out[((ob * 2 + 1) * H_ + yy) * W_ + xx] = o1;
        }
    }
}

// ---------------------------------------------------------------------------

extern "C" void kernel_launch(void* const* d_in, const int* in_sizes, int n_in,
                              void* d_out, int out_size, void* d_ws, size_t ws_size,
                              hipStream_t stream) {
    const float* feature   = (const float*)d_in[0];
    const float* init_flow = (const float*)d_in[1];
    const float* flow_fix  = (const float*)d_in[2];
    const float* w1        = (const float*)d_in[3];
    const float* b1        = (const float*)d_in[4];
    const float* prelu_w   = (const float*)d_in[5];
    const float* w2        = (const float*)d_in[6];
    const float* b2        = (const float*)d_in[7];

    _Float16* Wt1 = (_Float16*)d_ws;                       // 256*4704*2 = 2.30 MB
    _Float16* Wt2 = Wt1 + (size_t)HID * KPAD;              // 48*256*2  = 24 KB

    pack_w1_kernel<<<(HID * KPAD + 255) / 256, 256, 0, stream>>>(w1, Wt1);
    pack_w2_kernel<<<(N2 * HID + 255) / 256, 256, 0, stream>>>(w2, Wt2);

    dim3 grid(MTOT / TM);   // 1792 workgroups
    merging_main_kernel<<<grid, 256, LDS_TOTAL, stream>>>(
        feature, init_flow, flow_fix, b1, prelu_w, b2, Wt1, Wt2, (float*)d_out);
}
